// LSTMBiA_21672404976312
// MI455X (gfx1250) — matrix operand order repaired and checked by reading, not compiled, not run on hardware
//
#include <hip/hip_runtime.h>
#include <hip/hip_bf16.h>
#include <math.h>

// ---------------------------------------------------------------------------
// Types for WMMA fragments (CDNA5 gfx1250, wave32)
// ---------------------------------------------------------------------------
typedef __attribute__((ext_vector_type(16))) __bf16        v16bf;
typedef __attribute__((ext_vector_type(8)))  float         v8f;
typedef __attribute__((ext_vector_type(4)))  unsigned int  u32x4;

union Frag16 {
    v16bf bf;
    u32x4 u4[2];
};

static __device__ __forceinline__ unsigned short f2bf(float f) {
    unsigned int u = __float_as_uint(f);
    u = (u + 0x7FFFu + ((u >> 16) & 1u)) >> 16;   // round-to-nearest-even
    return (unsigned short)u;
}

static __device__ __forceinline__ float sigm(float x) {
    return 1.0f / (1.0f + expf(-x));
}

// ---------------------------------------------------------------------------
// Weight / activation conversion: f32 [R,C] -> bf16 [R,Cp] (zero padded)
// ---------------------------------------------------------------------------
__global__ void convert_pad_bf16(const float* __restrict__ src,
                                 unsigned short* __restrict__ dst,
                                 int R, int C, int Cp) {
    long long gid = (long long)blockIdx.x * 256 + threadIdx.x;
    long long total = (long long)R * Cp;
    if (gid >= total) return;
    int r = (int)(gid / Cp), c = (int)(gid % Cp);
    float v = (c < C) ? src[(long long)r * C + c] : 0.0f;
    dst[gid] = f2bf(v);
}

// ---------------------------------------------------------------------------
// Sequence-length computation for the 1024 sentence-level sequences
// rows 0..511  : conv turns (b*32+c), len = (c < conv_lens[b]) ? turn_len : 0
// rows 512..1023: user turns (b*32+u), len = (u < user_lens[b]) ? nnz : 0
// ---------------------------------------------------------------------------
__global__ void compute_slens(const int* __restrict__ conv_lens,
                              const int* __restrict__ conv_turn_lens,
                              const int* __restrict__ users,
                              const int* __restrict__ user_lens,
                              int* __restrict__ slens) {
    int idx = blockIdx.x * 256 + threadIdx.x;
    if (idx >= 1024) return;
    if (idx < 512) {
        int b = idx >> 5, c = idx & 31;
        slens[idx] = (c < conv_lens[b]) ? conv_turn_lens[idx] : 0;
    } else {
        int r = idx - 512;
        int b = r >> 5, u = r & 31;
        const int* tok = users + r * 50;
        int cnt = 0;
        for (int w = 0; w < 50; ++w) cnt += (tok[w] != 0) ? 1 : 0;
        slens[idx] = (u < user_lens[b]) ? cnt : 0;
    }
}

// ---------------------------------------------------------------------------
// Embedding gather -> bf16 [1024*50, 320]  (rows 0..25599 convs, rest users)
// ---------------------------------------------------------------------------
__global__ void gather_embed(const float* __restrict__ emb,
                             const int* __restrict__ convs,
                             const int* __restrict__ users,
                             unsigned short* __restrict__ out) {
    long long gid = (long long)blockIdx.x * 256 + threadIdx.x;
    long long total = (long long)51200 * 320;
    if (gid >= total) return;
    int row = (int)(gid / 320);
    int col = (int)(gid % 320);
    int tok = (row < 25600) ? convs[row] : users[row - 25600];
    float v = (col < 300) ? emb[(long long)tok * 300 + col] : 0.0f;
    out[gid] = f2bf(v);
}

// ---------------------------------------------------------------------------
// Generic WMMA bf16 GEMM:  C[M,N] = A[M,K] * B[N,K]^T + bias[N]   (row major)
// K multiple of 32, M multiple of 32, N multiple of 256.
// Block = 128 threads (4 waves). Block tile = 32x256; each wave owns a 32x64
// tile: 2 A fragments reused across 4 N-tiles (8 WMMAs per k-step), giving
// ~21 flops per loaded byte instead of 8 for the naive 16x16 scheme.
// Fragment layouts per CDNA5 ISA 7.12.2 (wave32):
//   A 16x32 bf16 : lane->M=lane&15 ; K = (lane<16?0:8) + {0..7, 16..23}
//   B 32x16 bf16 : lane->N=lane&15 ; K = (lane<16?0:8) + {0..7, 16..23}
//   C 16x16 f32  : lane->N=lane&15 ; elem v -> M = v + (lane<16?0:8)
// ---------------------------------------------------------------------------
__global__ __launch_bounds__(128)
void gemm_bf16_nt(const unsigned short* __restrict__ A, int lda,
                  const unsigned short* __restrict__ B, int ldb,
                  const float* __restrict__ bias,
                  float* __restrict__ C, int ldc,
                  unsigned short* __restrict__ Cbf, int ldcb,
                  int M, int N, int K, int relu) {
    int wave = threadIdx.x >> 5;
    int lane = threadIdx.x & 31;
    int m0 = blockIdx.x * 32;
    int n0 = blockIdx.y * 256 + wave * 64;
    if (m0 >= M || n0 >= N) return;
    int ln = lane & 15;
    int hi = lane >> 4;

    v8f acc[4][2];
#pragma unroll
    for (int nt = 0; nt < 4; ++nt) {
        float bv = bias ? bias[n0 + nt * 16 + ln] : 0.0f;
#pragma unroll
        for (int mt = 0; mt < 2; ++mt)
#pragma unroll
            for (int v = 0; v < 8; ++v) acc[nt][mt][v] = bv;
    }

    const unsigned short* arow0 = A + (long long)(m0 + ln) * lda + hi * 8;
    const unsigned short* arow1 = A + (long long)(m0 + 16 + ln) * lda + hi * 8;
    for (int k = 0; k < K; k += 32) {
        Frag16 fa0, fa1;
        fa0.u4[0] = *(const u32x4*)(arow0 + k);
        fa0.u4[1] = *(const u32x4*)(arow0 + k + 16);
        fa1.u4[0] = *(const u32x4*)(arow1 + k);
        fa1.u4[1] = *(const u32x4*)(arow1 + k + 16);
#pragma unroll
        for (int nt = 0; nt < 4; ++nt) {
            Frag16 fb;
            const unsigned short* bp =
                B + (long long)(n0 + nt * 16 + ln) * ldb + hi * 8 + k;
            fb.u4[0] = *(const u32x4*)(bp);
            fb.u4[1] = *(const u32x4*)(bp + 16);
            acc[nt][0] = __builtin_amdgcn_wmma_f32_16x16x32_bf16(
                false, fa0.bf, false, fb.bf, (short)0, acc[nt][0], false, false);
            acc[nt][1] = __builtin_amdgcn_wmma_f32_16x16x32_bf16(
                false, fa1.bf, false, fb.bf, (short)0, acc[nt][1], false, false);
        }
    }

#pragma unroll
    for (int nt = 0; nt < 4; ++nt) {
        int n = n0 + nt * 16 + ln;
#pragma unroll
        for (int mt = 0; mt < 2; ++mt) {
#pragma unroll
            for (int v = 0; v < 8; ++v) {
                int m = m0 + mt * 16 + v + hi * 8;
                float x = acc[nt][mt][v];
                if (relu) x = x > 0.0f ? x : 0.0f;
                C[(long long)m * ldc + n] = x;
                if (Cbf) Cbf[(long long)m * ldcb + n] = f2bf(x);
            }
        }
    }
}

// ---------------------------------------------------------------------------
// LSTM recurrence (one direction). Hidden = 256, gates = 1024 (i,f,g,o).
// Workgroup: 16 sequences, 256 threads (8 waves x 128 gate columns).
// Per step: g = xg[:,t,:] + h_bf16 @ Whh^T  via WMMA, then f32 gate update.
// Dynamic LDS: h_bf16 (8KB) + gates (64KB) + c (16KB) + h_f32 (16KB) = 104KB.
// ---------------------------------------------------------------------------
__global__ __launch_bounds__(256)
void lstm_scan(const float* __restrict__ xg,            // [NS, T, 1024]
               const unsigned short* __restrict__ Whh,  // [1024, 256] bf16
               const int* __restrict__ slens,           // [NS]
               int T, int reverse,
               float* __restrict__ outs, int outLd, int outOff,  // or null
               float* __restrict__ hlast, int hlLd, int hlOff)   // or null
{
    extern __shared__ char smem[];
    unsigned short* h_bf = (unsigned short*)smem;                 // 16*256
    float* gbuf = (float*)(smem + 8192);                          // 16*1024
    float* cbuf = (float*)(smem + 8192 + 65536);                  // 16*256
    float* hbuf = (float*)(smem + 8192 + 65536 + 16384);          // 16*256

    int tid = threadIdx.x;
    int s0  = blockIdx.x * 16;
    for (int i = tid; i < 16 * 256; i += 256) {
        h_bf[i] = 0; cbuf[i] = 0.0f; hbuf[i] = 0.0f;
    }
    __syncthreads();

    int wave = tid >> 5, lane = tid & 31, ln = lane & 15, hi = lane >> 4;
    int nbase = wave * 128 + ln;

    for (int tt = 0; tt < T; ++tt) {
        int t = reverse ? (T - 1 - tt) : tt;

        v8f acc[8];
#pragma unroll
        for (int nt = 0; nt < 8; ++nt) {
            int ncol = nbase + nt * 16;
#pragma unroll
            for (int v = 0; v < 8; ++v) {
                int m = v + hi * 8;
                acc[nt][v] = xg[((long long)(s0 + m) * T + t) * 1024 + ncol];
            }
        }
        const unsigned short* arow = h_bf + ln * 256 + hi * 8;
        for (int k = 0; k < 256; k += 32) {
            Frag16 fa;
            fa.u4[0] = *(const u32x4*)(arow + k);
            fa.u4[1] = *(const u32x4*)(arow + k + 16);
#pragma unroll
            for (int nt = 0; nt < 8; ++nt) {
                int ncol = nbase + nt * 16;
                Frag16 fb;
                const unsigned short* bp = Whh + (long long)ncol * 256 + hi * 8 + k;
                fb.u4[0] = *(const u32x4*)(bp);
                fb.u4[1] = *(const u32x4*)(bp + 16);
                acc[nt] = __builtin_amdgcn_wmma_f32_16x16x32_bf16(
                    false, fa.bf, false, fb.bf, (short)0, acc[nt], false, false);
            }
        }
#pragma unroll
        for (int nt = 0; nt < 8; ++nt) {
            int ncol = nbase + nt * 16;
#pragma unroll
            for (int v = 0; v < 8; ++v) {
                int m = v + hi * 8;
                gbuf[m * 1024 + ncol] = acc[nt][v];
            }
        }
        __syncthreads();

        // elementwise gate update, masking, output
        for (int idx = tid; idx < 16 * 256; idx += 256) {
            int s = idx >> 8, j = idx & 255;
            float gi = gbuf[s * 1024 + j];
            float gf = gbuf[s * 1024 + 256 + j];
            float gg = gbuf[s * 1024 + 512 + j];
            float go = gbuf[s * 1024 + 768 + j];
            float c = cbuf[idx], h = hbuf[idx];
            float cn = sigm(gf) * c + sigm(gi) * tanhf(gg);
            float hn = sigm(go) * tanhf(cn);
            bool msk = t < slens[s0 + s];
            c = msk ? cn : c;
            h = msk ? hn : h;
            cbuf[idx] = c;
            hbuf[idx] = h;
            h_bf[idx] = f2bf(h);
            if (outs)
                outs[((long long)(s0 + s) * T + t) * outLd + outOff + j] =
                    msk ? hn : 0.0f;
        }
        __syncthreads();
    }

    if (hlast) {
        for (int idx = tid; idx < 16 * 256; idx += 256) {
            int s = idx >> 8, j = idx & 255;
            hlast[(long long)(s0 + s) * hlLd + hlOff + j] = hbuf[idx];
        }
    }
}

// ---------------------------------------------------------------------------
// Attention / similarity / feature construction (per batch b).
// conv_out  f32 [512,512] row = b*32+i ; his f32 [512,512] row = b*32+j.
// feat bf16 [512, 2048].
// ---------------------------------------------------------------------------
__global__ __launch_bounds__(256)
void attention_feat(const float* __restrict__ conv_out,
                    const float* __restrict__ his,
                    const float* __restrict__ sim_w,
                    const float* __restrict__ sim_b,
                    unsigned short* __restrict__ feat) {
    __shared__ float P[32][32];
    __shared__ float ci_s[32], hj_s[32], rmax_s[32], aw[32];
    __shared__ float ah2c[512];

    int b = blockIdx.x;
    int tid = threadIdx.x;
    const float* wc = sim_w;
    const float* wh = sim_w + 512;
    const float* wm = sim_w + 1024;
    float sb = sim_b[0];
    const float* CO = conv_out + (long long)b * 32 * 512;
    const float* HO = his + (long long)b * 32 * 512;

    if (tid < 64) {
        int i = tid & 31;
        const float* base = (tid < 32) ? (CO + i * 512) : (HO + i * 512);
        const float* w = (tid < 32) ? wc : wh;
        float s = 0.0f;
        for (int d = 0; d < 512; ++d) s += base[d] * w[d];
        if (tid < 32) ci_s[i] = s; else hj_s[i] = s;
    }
    __syncthreads();

    for (int p = tid; p < 1024; p += 256) {
        int i = p >> 5, j = p & 31;
        const float* a = CO + i * 512;
        const float* h = HO + j * 512;
        float s = 0.0f;
        for (int d = 0; d < 512; ++d) s += a[d] * wm[d] * h[d];
        P[i][j] = ci_s[i] + hj_s[j] + s + sb;
    }
    __syncthreads();

    if (tid < 32) {
        int i = tid;
        float mx = P[i][0];
        for (int j = 1; j < 32; ++j) mx = fmaxf(mx, P[i][j]);
        rmax_s[i] = mx;
        float sum = 0.0f;
        for (int j = 0; j < 32; ++j) { float e = expf(P[i][j] - mx); P[i][j] = e; sum += e; }
        float inv = 1.0f / sum;
        for (int j = 0; j < 32; ++j) P[i][j] *= inv;
    }
    __syncthreads();

    if (tid == 0) {
        float mx = rmax_s[0];
        for (int i = 1; i < 32; ++i) mx = fmaxf(mx, rmax_s[i]);
        float sum = 0.0f;
        for (int i = 0; i < 32; ++i) { float e = expf(rmax_s[i] - mx); aw[i] = e; sum += e; }
        float inv = 1.0f / sum;
        for (int i = 0; i < 32; ++i) aw[i] *= inv;
    }
    __syncthreads();

    for (int d = tid; d < 512; d += 256) {
        float s = 0.0f;
        for (int i = 0; i < 32; ++i) s += aw[i] * CO[i * 512 + d];
        ah2c[d] = s;
    }
    __syncthreads();

    for (int p = tid; p < 32 * 512; p += 256) {
        int i = p >> 9, d = p & 511;
        float co = CO[i * 512 + d];
        float ac = 0.0f;
        for (int j = 0; j < 32; ++j) ac += P[i][j] * HO[j * 512 + d];
        unsigned short* fr = feat + ((long long)(b * 32 + i)) * 2048;
        fr[d]        = f2bf(co);
        fr[512 + d]  = f2bf(ac);
        fr[1024 + d] = f2bf(co * ac);
        fr[1536 + d] = f2bf(co * ah2c[d]);
    }
}

// ---------------------------------------------------------------------------
// Final head: labels[b] = sigmoid(model_out[b] . out_W + out_b)
// model_out already lives at dout+16.
// ---------------------------------------------------------------------------
__global__ void final_head(const float* __restrict__ out_W,
                           const float* __restrict__ out_b,
                           float* __restrict__ dout) {
    int tid = threadIdx.x;
    if (tid < 16) {
        const float* mo = dout + 16 + tid * 512;
        float s = out_b[0];
        for (int d = 0; d < 512; ++d) s += mo[d] * out_W[d];
        dout[tid] = 1.0f / (1.0f + expf(-s));
    }
}

// ---------------------------------------------------------------------------
// Host orchestration
// ---------------------------------------------------------------------------
static inline size_t wsAlloc(size_t& off, size_t bytes) {
    size_t r = (off + 255) & ~(size_t)255;
    off = r + bytes;
    return r;
}

extern "C" void kernel_launch(void* const* d_in, const int* in_sizes, int n_in,
                              void* d_out, int out_size, void* d_ws, size_t ws_size,
                              hipStream_t stream) {
    (void)in_sizes; (void)n_in; (void)out_size; (void)ws_size;
    // --- input indices (params dict insertion order, then data tensors) ---
    const float* embedding = (const float*)d_in[0];
    // sent: 1..6, conv: 7..12, model1: 13..18, model2: 19..24
    const float* sWihF = (const float*)d_in[1];
    const float* sWhhF = (const float*)d_in[2];
    const float* sbF   = (const float*)d_in[3];
    const float* sWihB = (const float*)d_in[4];
    const float* sWhhB = (const float*)d_in[5];
    const float* sbB   = (const float*)d_in[6];
    const float* cWihF = (const float*)d_in[7];
    const float* cWhhF = (const float*)d_in[8];
    const float* cbF   = (const float*)d_in[9];
    const float* cWihB = (const float*)d_in[10];
    const float* cWhhB = (const float*)d_in[11];
    const float* cbB   = (const float*)d_in[12];
    const float* m1WihF = (const float*)d_in[13];
    const float* m1WhhF = (const float*)d_in[14];
    const float* m1bF   = (const float*)d_in[15];
    const float* m1WihB = (const float*)d_in[16];
    const float* m1WhhB = (const float*)d_in[17];
    const float* m1bB   = (const float*)d_in[18];
    const float* m2WihF = (const float*)d_in[19];
    const float* m2WhhF = (const float*)d_in[20];
    const float* m2bF   = (const float*)d_in[21];
    const float* m2WihB = (const float*)d_in[22];
    const float* m2WhhB = (const float*)d_in[23];
    const float* m2bB   = (const float*)d_in[24];
    const float* sim_w = (const float*)d_in[25];
    const float* sim_b = (const float*)d_in[26];
    const float* l1_W  = (const float*)d_in[27];
    const float* l1_b  = (const float*)d_in[28];
    const float* l2_W  = (const float*)d_in[29];
    const float* l2_b  = (const float*)d_in[30];
    const float* out_W = (const float*)d_in[31];
    const float* out_b = (const float*)d_in[32];
    const int* convs          = (const int*)d_in[33];
    const int* conv_lens      = (const int*)d_in[34];
    const int* conv_turn_lens = (const int*)d_in[35];
    const int* users          = (const int*)d_in[36];
    const int* user_lens      = (const int*)d_in[37];

    char* ws = (char*)d_ws;
    size_t off = 0;
    // activations
    unsigned short* embbf  = (unsigned short*)(ws + wsAlloc(off, (size_t)51200 * 320 * 2));
    float*          xg     = (float*)        (ws + wsAlloc(off, (size_t)51200 * 1024 * 4));
    float*          sentH  = (float*)        (ws + wsAlloc(off, (size_t)1024 * 512 * 4));
    unsigned short* sentHb = (unsigned short*)(ws + wsAlloc(off, (size_t)1024 * 512 * 2));
    int*            slens  = (int*)          (ws + wsAlloc(off, 1024 * 4));
    float*          convO  = (float*)        (ws + wsAlloc(off, (size_t)512 * 512 * 4));
    unsigned short* featb  = (unsigned short*)(ws + wsAlloc(off, (size_t)512 * 2048 * 2));
    float*          x1f    = (float*)        (ws + wsAlloc(off, (size_t)512 * 1024 * 4));
    unsigned short* x1b    = (unsigned short*)(ws + wsAlloc(off, (size_t)512 * 1024 * 2));
    float*          x2f    = (float*)        (ws + wsAlloc(off, (size_t)512 * 512 * 4));
    unsigned short* x2b    = (unsigned short*)(ws + wsAlloc(off, (size_t)512 * 512 * 2));
    float*          m1f    = (float*)        (ws + wsAlloc(off, (size_t)512 * 512 * 4));
    unsigned short* m1b    = (unsigned short*)(ws + wsAlloc(off, (size_t)512 * 512 * 2));
    // bf16 weights
    unsigned short* wsWihF = (unsigned short*)(ws + wsAlloc(off, (size_t)1024 * 320 * 2));
    unsigned short* wsWihB = (unsigned short*)(ws + wsAlloc(off, (size_t)1024 * 320 * 2));
    unsigned short* wsWhhF = (unsigned short*)(ws + wsAlloc(off, (size_t)1024 * 256 * 2));
    unsigned short* wsWhhB = (unsigned short*)(ws + wsAlloc(off, (size_t)1024 * 256 * 2));
    unsigned short* wcWihF = (unsigned short*)(ws + wsAlloc(off, (size_t)1024 * 512 * 2));
    unsigned short* wcWihB = (unsigned short*)(ws + wsAlloc(off, (size_t)1024 * 512 * 2));
    unsigned short* wcWhhF = (unsigned short*)(ws + wsAlloc(off, (size_t)1024 * 256 * 2));
    unsigned short* wcWhhB = (unsigned short*)(ws + wsAlloc(off, (size_t)1024 * 256 * 2));
    unsigned short* wm1WihF = (unsigned short*)(ws + wsAlloc(off, (size_t)1024 * 512 * 2));
    unsigned short* wm1WihB = (unsigned short*)(ws + wsAlloc(off, (size_t)1024 * 512 * 2));
    unsigned short* wm1WhhF = (unsigned short*)(ws + wsAlloc(off, (size_t)1024 * 256 * 2));
    unsigned short* wm1WhhB = (unsigned short*)(ws + wsAlloc(off, (size_t)1024 * 256 * 2));
    unsigned short* wm2WihF = (unsigned short*)(ws + wsAlloc(off, (size_t)1024 * 512 * 2));
    unsigned short* wm2WihB = (unsigned short*)(ws + wsAlloc(off, (size_t)1024 * 512 * 2));
    unsigned short* wm2WhhF = (unsigned short*)(ws + wsAlloc(off, (size_t)1024 * 256 * 2));
    unsigned short* wm2WhhB = (unsigned short*)(ws + wsAlloc(off, (size_t)1024 * 256 * 2));
    unsigned short* wl1 = (unsigned short*)(ws + wsAlloc(off, (size_t)1024 * 2048 * 2));
    unsigned short* wl2 = (unsigned short*)(ws + wsAlloc(off, (size_t)512 * 1024 * 2));

    auto cvt = [&](const float* s, unsigned short* d, int R, int C, int Cp) {
        long long n = (long long)R * Cp;
        convert_pad_bf16<<<dim3((unsigned)((n + 255) / 256)), 256, 0, stream>>>(s, d, R, C, Cp);
    };
    // --- weight conversion ---
    cvt(sWihF, wsWihF, 1024, 300, 320);  cvt(sWihB, wsWihB, 1024, 300, 320);
    cvt(sWhhF, wsWhhF, 1024, 256, 256);  cvt(sWhhB, wsWhhB, 1024, 256, 256);
    cvt(cWihF, wcWihF, 1024, 512, 512);  cvt(cWihB, wcWihB, 1024, 512, 512);
    cvt(cWhhF, wcWhhF, 1024, 256, 256);  cvt(cWhhB, wcWhhB, 1024, 256, 256);
    cvt(m1WihF, wm1WihF, 1024, 512, 512); cvt(m1WihB, wm1WihB, 1024, 512, 512);
    cvt(m1WhhF, wm1WhhF, 1024, 256, 256); cvt(m1WhhB, wm1WhhB, 1024, 256, 256);
    cvt(m2WihF, wm2WihF, 1024, 512, 512); cvt(m2WihB, wm2WihB, 1024, 512, 512);
    cvt(m2WhhF, wm2WhhF, 1024, 256, 256); cvt(m2WhhB, wm2WhhB, 1024, 256, 256);
    cvt(l1_W, wl1, 1024, 2048, 2048);
    cvt(l2_W, wl2, 512, 1024, 1024);

    compute_slens<<<4, 256, 0, stream>>>(conv_lens, conv_turn_lens, users, user_lens, slens);
    gather_embed<<<64000, 256, 0, stream>>>(embedding, convs, users, embbf);

    const size_t scanLds = 8192 + 65536 + 16384 + 16384;  // 104 KB

    auto gemm = [&](const unsigned short* A, int lda, const unsigned short* B, int ldb,
                    const float* bias, float* C, int ldc,
                    unsigned short* Cb, int ldcb, int M, int N, int K, int relu) {
        gemm_bf16_nt<<<dim3(M / 32, N / 256), 128, 0, stream>>>(
            A, lda, B, ldb, bias, C, ldc, Cb, ldcb, M, N, K, relu);
    };

    // --- sentence-level BiLSTM over 1024 sequences (conv turns + user turns)
    gemm(embbf, 320, wsWihF, 320, sbF, xg, 1024, nullptr, 0, 51200, 1024, 320, 0);
    lstm_scan<<<64, 256, scanLds, stream>>>(xg, wsWhhF, slens, 50, 0,
                                            nullptr, 0, 0, sentH, 512, 0);
    gemm(embbf, 320, wsWihB, 320, sbB, xg, 1024, nullptr, 0, 51200, 1024, 320, 0);
    lstm_scan<<<64, 256, scanLds, stream>>>(xg, wsWhhB, slens, 50, 1,
                                            nullptr, 0, 0, sentH, 512, 256);
    cvt(sentH, sentHb, 1024, 512, 512);

    // --- conv-level BiLSTM: 16 sequences of T=32, input = sentH rows 0..511
    gemm(sentHb, 512, wcWihF, 512, cbF, xg, 1024, nullptr, 0, 512, 1024, 512, 0);
    lstm_scan<<<1, 256, scanLds, stream>>>(xg, wcWhhF, conv_lens, 32, 0,
                                           convO, 512, 0, nullptr, 0, 0);
    gemm(sentHb, 512, wcWihB, 512, cbB, xg, 1024, nullptr, 0, 512, 1024, 512, 0);
    lstm_scan<<<1, 256, scanLds, stream>>>(xg, wcWhhB, conv_lens, 32, 1,
                                           convO, 512, 256, nullptr, 0, 0);

    // --- attention + feature construction (his_out = sentH rows 512..1023)
    attention_feat<<<16, 256, 0, stream>>>(convO, sentH + (size_t)512 * 512,
                                           sim_w, sim_b, featb);

    // --- l1 / l2 dense + relu
    gemm(featb, 2048, wl1, 2048, l1_b, x1f, 1024, x1b, 1024, 512, 1024, 2048, 1);
    gemm(x1b, 1024, wl2, 1024, l2_b, x2f, 512, x2b, 512, 512, 512, 1024, 1);

    // --- model1 BiLSTM (outs needed)
    gemm(x2b, 512, wm1WihF, 512, m1bF, xg, 1024, nullptr, 0, 512, 1024, 512, 0);
    lstm_scan<<<1, 256, scanLds, stream>>>(xg, wm1WhhF, conv_lens, 32, 0,
                                           m1f, 512, 0, nullptr, 0, 0);
    gemm(x2b, 512, wm1WihB, 512, m1bB, xg, 1024, nullptr, 0, 512, 1024, 512, 0);
    lstm_scan<<<1, 256, scanLds, stream>>>(xg, wm1WhhB, conv_lens, 32, 1,
                                           m1f, 512, 256, nullptr, 0, 0);
    cvt(m1f, m1b, 512, 512, 512);

    // --- model2 BiLSTM (h_last -> d_out+16 directly)
    float* dout = (float*)d_out;
    gemm(m1b, 512, wm2WihF, 512, m2bF, xg, 1024, nullptr, 0, 512, 1024, 512, 0);
    lstm_scan<<<1, 256, scanLds, stream>>>(xg, wm2WhhF, conv_lens, 32, 0,
                                           nullptr, 0, 0, dout + 16, 512, 0);
    gemm(m1b, 512, wm2WihB, 512, m2bB, xg, 1024, nullptr, 0, 512, 1024, 512, 0);
    lstm_scan<<<1, 256, scanLds, stream>>>(xg, wm2WhhB, conv_lens, 32, 1,
                                           nullptr, 0, 0, dout + 16, 512, 256);

    final_head<<<1, 256, 0, stream>>>(out_W, out_b, dout);
}
